// Decode_79577154060436
// MI455X (gfx1250) — matrix operand order repaired and checked by reading, not compiled, not run on hardware
//
#include <hip/hip_runtime.h>
#include <hip/hip_bf16.h>

// ---------------- vector typedefs (WMMA operand shapes, wave32) --------------
typedef __attribute__((ext_vector_type(16))) __bf16 v16bf;
typedef __attribute__((ext_vector_type(8)))  __bf16 v8bf;
typedef __attribute__((ext_vector_type(4)))  __bf16 bf4;
typedef __attribute__((ext_vector_type(8)))  float  v8f;
typedef __attribute__((ext_vector_type(4)))  float  f4;

#define GN_EPS 1e-5f

// LDS / ws blob layout constants (bytes)
// Edge blob: Wc1t[128][264]bf16 | Wc2t[128][136]bf16 | W_rp f32[512] | b_rp[128] | g_c1[128] | b_c1[128]
#define OFF_WC2   67584      // 128*264*2
#define OFF_WRP   102400     // +128*136*2
#define OFF_BRP   104448
#define OFF_GC1   104960
#define OFF_BC1   105472
#define S1        105984     // edge blob size
#define LK1       264        // padded K stride for Wc1t / edge A buffer
#define LK2       136        // padded K stride for 128-K weights / 16x128 buffers
#define BUFA1     (16*LK1)   // per-wave A-stage elements (16x256, edge)
#define BUFA2     (16*LK2)   // per-wave 16x128 buffer elements
// edge: blob + 8x bufA(16x256) + 8x bufB(16x128)
#define SMEM_EDGE (S1 + 8*BUFA1*2 + 8*BUFA2*2)   // 208384 B
// Node blob: Wint | Wm1t | Wm2t | g_n b_n g_m1 bm1 g_m2 bm2
#define OFF_WM1   34816
#define OFF_WM2   69632
#define OFF_GN    104448
#define OFF_BN    104960
#define OFF_GM1   105472
#define OFF_BM1   105984
#define OFF_GM2   106496
#define OFF_BM2   107008
#define S2        107520
// node: blob + 3 ping-pong 16x128 buffers per wave (A/B/C rotation)
#define SMEM_NODE (S2 + 24*BUFA2*2)              // 211968 B

// ---------------- helpers ----------------------------------------------------
__device__ __forceinline__ v16bf ldfrag(const __bf16* p, const __bf16* q) {
    v8bf lo = *(const v8bf*)p;
    v8bf hi = *(const v8bf*)q;
    return __builtin_shufflevector(lo, hi, 0,1,2,3,4,5,6,7,8,9,10,11,12,13,14,15);
}

__device__ __forceinline__ void zero8(v8f c[8]) {
#pragma unroll
    for (int t = 0; t < 8; ++t)
#pragma unroll
        for (int j = 0; j < 8; ++j) c[t][j] = 0.0f;
}

// 16xK (A in LDS, bf16) x Kx128 (Wt in LDS, [n][k] bf16) -> c[8] f32 accums.
// All 8 B-fragments of a k-chunk are batch-loaded first so the WMMA chain
// issues against staggered s_wait_dscnt counts instead of waiting per-op.
// (Requires the relaxed VGPR cap from __launch_bounds__(256,2).)
template<int NKC, int LKW>
__device__ __forceinline__ void mm_tile(const __bf16* bufA, int lda,
                                        const __bf16* W, int ln, int hf, v8f c[8]) {
    const __bf16* Arow = bufA + ln * lda + 16 * hf;
    const __bf16* Wrow = W + ln * LKW + 16 * hf;
#pragma unroll
    for (int kc = 0; kc < NKC; ++kc) {
        const int k0 = kc * 32;
        v16bf a = ldfrag(Arow + k0, Arow + k0 + 8);
        v16bf b[8];
#pragma unroll
        for (int t = 0; t < 8; ++t) {
            const __bf16* bp = Wrow + t * (16 * LKW) + k0;
            b[t] = ldfrag(bp, bp + 8);
        }
#pragma unroll
        for (int t = 0; t < 8; ++t)
            c[t] = __builtin_amdgcn_wmma_f32_16x16x32_bf16(
                false, a, false, b[t], (short)0, c[t], false, false);
    }
}

// Row mean / inv-std for 16x128 tile held in c[8] accumulators.
// Row m lives in VGPR j=m%8 across the 16-lane half (l>>4)==(m/8), so 4 xor
// shuffles (masks 1,2,4,8) reduce each half independently.
__device__ __forceinline__ void gn_stats(const v8f c[8], float mean[8], float inv[8]) {
    float ps[8], pq[8];
#pragma unroll
    for (int j = 0; j < 8; ++j) { ps[j] = 0.f; pq[j] = 0.f; }
#pragma unroll
    for (int t = 0; t < 8; ++t)
#pragma unroll
        for (int j = 0; j < 8; ++j) { float v = c[t][j]; ps[j] += v; pq[j] += v * v; }
#pragma unroll
    for (int m = 1; m < 16; m <<= 1)
#pragma unroll
        for (int j = 0; j < 8; ++j) {
            ps[j] += __shfl_xor(ps[j], m, 32);
            pq[j] += __shfl_xor(pq[j], m, 32);
        }
#pragma unroll
    for (int j = 0; j < 8; ++j) {
        float mu = ps[j] * (1.0f / 128.0f);
        float var = pq[j] * (1.0f / 128.0f) - mu * mu;
        mean[j] = mu;
        inv[j] = rsqrtf(var + GN_EPS);
    }
}

__device__ __forceinline__ void gn_store_bf16(const v8f c[8], const float* g,
                                              const float* b, __bf16* dst, int lda,
                                              int ln, int hf, bool relu) {
    float mean[8], inv[8];
    gn_stats(c, mean, inv);
#pragma unroll
    for (int t = 0; t < 8; ++t) {
        float gg = g[t * 16 + ln], bb = b[t * 16 + ln];
#pragma unroll
        for (int j = 0; j < 8; ++j) {
            float v = (c[t][j] - mean[j]) * inv[j] * gg + bb;
            if (relu) v = fmaxf(v, 0.f);
            dst[(j + 8 * hf) * lda + t * 16 + ln] = (__bf16)v;
        }
    }
}

// ---------------- prep kernels ----------------------------------------------
__global__ void zero_f32(float* p, int n) {
    int i = blockIdx.x * blockDim.x + threadIdx.x;
    if (i < n) p[i] = 0.0f;
}

__global__ void copy_f32(const float* __restrict__ s, float* __restrict__ d, int n) {
    int i = blockIdx.x * blockDim.x + threadIdx.x;
    if (i < n) d[i] = s[i];
}

// Wt[n*LK + k] = (bf16) W[k*N + n], pad columns zeroed
__global__ void transpose_bf16(const float* __restrict__ W, __bf16* __restrict__ Wt,
                               int K, int N, int LK) {
    int idx = blockIdx.x * blockDim.x + threadIdx.x;
    if (idx >= N * LK) return;
    int n = idx / LK, k = idx % LK;
    float v = (k < K) ? W[k * N + n] : 0.0f;
    Wt[idx] = (__bf16)v;
}

// ---------------- edge kernel ------------------------------------------------
// Per 16-edge tile (one wave): gather ctx rows + relpose linear -> A[16x256]bf16
// -> WMMA (256->128) -> GN+ReLU (in-register stats) -> bufB -> WMMA (128->128)
// -> f32 atomic scatter into acc[wi].  Ping-pong buffers => 2 barriers/tile.
// launch_bounds(256,2): true occupancy is LDS-bound at 2 waves/SIMD, so let
// the compiler use a large VGPR budget for deep WMMA/ds_load pipelining.
__global__ __launch_bounds__(256, 2)
void edge_kernel(const float* __restrict__ ctxf, const float* __restrict__ cpose,
                 const float* __restrict__ tpose, const int* __restrict__ hi,
                 const int* __restrict__ wi, const char* __restrict__ blob,
                 float* __restrict__ acc, int E, int ntiles, int iters) {
    extern __shared__ char smem[];
    __bf16* Wc1 = (__bf16*)smem;
    __bf16* Wc2 = (__bf16*)(smem + OFF_WC2);
    const float* Wrp = (const float*)(smem + OFF_WRP);
    const float* brp = (const float*)(smem + OFF_BRP);
    const float* gc1 = (const float*)(smem + OFF_GC1);
    const float* bc1 = (const float*)(smem + OFF_BC1);
    const int wib = threadIdx.x >> 5;
    __bf16* bufA = (__bf16*)(smem + S1) + wib * BUFA1;
    __bf16* bufB = (__bf16*)(smem + S1 + 8 * BUFA1 * 2) + wib * BUFA2;

    { // stage weight blob (flat copy, 16B chunks)
        const int4* src = (const int4*)blob;
        int4* dst = (int4*)smem;
        for (int i = threadIdx.x; i < S1 / 16; i += blockDim.x) dst[i] = src[i];
    }
    __syncthreads();

    const int lane = threadIdx.x & 31;
    const int ln = lane & 15;
    const int hf = lane >> 4;
    const int wave = blockIdx.x * (blockDim.x >> 5) + wib;
    const int nw = gridDim.x * (blockDim.x >> 5);

    for (int it = 0; it < iters; ++it) {
        const int tile = wave + it * nw;
        const bool active = tile < ntiles;   // wave-uniform -> EXEC stays all-ones
        const int e0 = tile * 16;

        v8f c[8];
        if (active) {
            // ---- stage A[16x256]: lane handles row ln, 64-col half hf ----
            int e = e0 + ln;
            bool ev = e < E;
            int hidx = ev ? hi[e] : 0;
            int widx = ev ? wi[e] : 0;
            const int cb0 = 64 * hf;
            const float* srow = ctxf + (size_t)hidx * 128 + cb0;
            __bf16* drow = bufA + ln * LK1;
#pragma unroll
            for (int j = 0; j < 16; ++j) {
                f4 v = *(const f4*)(srow + 4 * j);
                bf4 o;
#pragma unroll
                for (int q = 0; q < 4; ++q) o[q] = (__bf16)v[q];
                *(bf4*)(drow + cb0 + 4 * j) = o;
            }
            // relpose: relu((cp[hi]-tp[wi]) @ W_rp + b_rp) -> cols 128..255
            float d0 = cpose[hidx * 4 + 0] - tpose[widx * 4 + 0];
            float d1 = cpose[hidx * 4 + 1] - tpose[widx * 4 + 1];
            float d2 = cpose[hidx * 4 + 2] - tpose[widx * 4 + 2];
            float d3 = cpose[hidx * 4 + 3] - tpose[widx * 4 + 3];
#pragma unroll
            for (int j = 0; j < 16; ++j) {
                int cb = cb0 + 4 * j;
                f4 w0 = *(const f4*)(Wrp + cb);
                f4 w1 = *(const f4*)(Wrp + 128 + cb);
                f4 w2 = *(const f4*)(Wrp + 256 + cb);
                f4 w3 = *(const f4*)(Wrp + 384 + cb);
                f4 bb = *(const f4*)(brp + cb);
                bf4 o;
#pragma unroll
                for (int q = 0; q < 4; ++q) {
                    float v = fmaf(d0, w0[q], fmaf(d1, w1[q],
                              fmaf(d2, w2[q], fmaf(d3, w3[q], bb[q]))));
                    o[q] = (__bf16)fmaxf(v, 0.f);
                }
                *(bf4*)(drow + 128 + cb) = o;
            }
        }
        __syncthreads();                     // bufA visible to all lanes

        if (active) {
            zero8(c);
            mm_tile<8, LK1>(bufA, LK1, Wc1, ln, hf, c);   // [16x256]@[256x128]
            gn_store_bf16(c, gc1, bc1, bufB, LK2, ln, hf, true);
        }
        __syncthreads();                     // bufB visible to all lanes

        if (active) {
            zero8(c);
            mm_tile<4, LK2>(bufB, LK2, Wc2, ln, hf, c);   // [16x128]@[128x128]
            // ---- scatter: acc[wi[e]][col] += c ----
#pragma unroll
            for (int j = 0; j < 8; ++j) {
                int e = e0 + j + 8 * hf;
                int w = (e < E) ? wi[e] : -1;
                if (w >= 0) {
                    float* arow = acc + (size_t)w * 128 + ln;
#pragma unroll
                    for (int t = 0; t < 8; ++t)
                        unsafeAtomicAdd(arow + t * 16, c[t][j]);
                }
            }
        }
        // no loop-end barrier: next stage writes bufA, whose last readers
        // (mm1) precede the bufB barrier; next bufB writes follow next barrier.
    }
}

// ---------------- node kernel ------------------------------------------------
// tf = tgt@W_input + acc; relu(GN); relu(GN(tf@W_m1)); GN(tf@W_m2); relu(+tgt)
// Buffer rotation A -> B -> C keeps it at 3 barriers per tile.
__global__ __launch_bounds__(256, 2)
void node_kernel(const float* __restrict__ tgtf, const float* __restrict__ acc,
                 const char* __restrict__ blob, float* __restrict__ out,
                 int Ntgt, int ntiles, int iters) {
    extern __shared__ char smem[];
    __bf16* Wi  = (__bf16*)smem;
    __bf16* Wm1 = (__bf16*)(smem + OFF_WM1);
    __bf16* Wm2 = (__bf16*)(smem + OFF_WM2);
    const float* gn  = (const float*)(smem + OFF_GN);
    const float* bn  = (const float*)(smem + OFF_BN);
    const float* gm1 = (const float*)(smem + OFF_GM1);
    const float* bm1 = (const float*)(smem + OFF_BM1);
    const float* gm2 = (const float*)(smem + OFF_GM2);
    const float* bm2 = (const float*)(smem + OFF_BM2);
    const int wib = threadIdx.x >> 5;
    __bf16* bufA = (__bf16*)(smem + S2) + wib * BUFA2;
    __bf16* bufB = (__bf16*)(smem + S2 + 8 * BUFA2 * 2) + wib * BUFA2;
    __bf16* bufC = (__bf16*)(smem + S2 + 16 * BUFA2 * 2) + wib * BUFA2;

    {
        const int4* src = (const int4*)blob;
        int4* dst = (int4*)smem;
        for (int i = threadIdx.x; i < S2 / 16; i += blockDim.x) dst[i] = src[i];
    }
    __syncthreads();

    const int lane = threadIdx.x & 31;
    const int ln = lane & 15;
    const int hf = lane >> 4;
    const int wave = blockIdx.x * (blockDim.x >> 5) + wib;
    const int nw = gridDim.x * (blockDim.x >> 5);

    for (int it = 0; it < iters; ++it) {
        const int tile = wave + it * nw;
        const bool active = tile < ntiles;
        const int r0 = tile * 16;

        v8f c[8];
        if (active) {
            int r = r0 + ln;
            const float* srow = tgtf + (size_t)((r < Ntgt) ? r : 0) * 128 + 64 * hf;
            __bf16* drow = bufA + ln * LK2 + 64 * hf;
#pragma unroll
            for (int j = 0; j < 16; ++j) {
                f4 v = *(const f4*)(srow + 4 * j);
                bf4 o;
#pragma unroll
                for (int q = 0; q < 4; ++q) o[q] = (__bf16)v[q];
                *(bf4*)(drow + 4 * j) = o;
            }
        }
        __syncthreads();                     // bufA ready

        if (active) {
            zero8(c);
            mm_tile<4, LK2>(bufA, LK2, Wi, ln, hf, c);    // tgt @ W_input
#pragma unroll
            for (int t = 0; t < 8; ++t)
#pragma unroll
                for (int j = 0; j < 8; ++j) {
                    int gr = r0 + j + 8 * hf;
                    if (gr < Ntgt) c[t][j] += acc[(size_t)gr * 128 + t * 16 + ln];
                }
            gn_store_bf16(c, gn, bn, bufB, LK2, ln, hf, true);
        }
        __syncthreads();                     // bufB ready

        if (active) {
            zero8(c);
            mm_tile<4, LK2>(bufB, LK2, Wm1, ln, hf, c);
            gn_store_bf16(c, gm1, bm1, bufC, LK2, ln, hf, true);
        }
        __syncthreads();                     // bufC ready

        if (active) {
            zero8(c);
            mm_tile<4, LK2>(bufC, LK2, Wm2, ln, hf, c);
            float mean[8], inv[8];
            gn_stats(c, mean, inv);
#pragma unroll
            for (int t = 0; t < 8; ++t) {
                int col = t * 16 + ln;
                float gg = gm2[col], bb = bm2[col];
#pragma unroll
                for (int j = 0; j < 8; ++j) {
                    int gr = r0 + j + 8 * hf;
                    if (gr < Ntgt) {
                        float v = (c[t][j] - mean[j]) * inv[j] * gg + bb;
                        float res = tgtf[(size_t)gr * 128 + col];
                        out[(size_t)gr * 128 + col] = fmaxf(v + res, 0.f);
                    }
                }
            }
        }
        // no loop-end barrier: next stage writes bufA, last read before sync2.
    }
}

// ---------------- host -------------------------------------------------------
extern "C" void kernel_launch(void* const* d_in, const int* in_sizes, int n_in,
                              void* d_out, int out_size, void* d_ws, size_t ws_size,
                              hipStream_t stream) {
    const float* ctx_feat = (const float*)d_in[0];
    const float* tgt_feat = (const float*)d_in[1];
    const float* cpose    = (const float*)d_in[2];
    const float* tpose    = (const float*)d_in[3];
    const float* W_input  = (const float*)d_in[4];
    const float* W_rp     = (const float*)d_in[5];
    const float* b_rp     = (const float*)d_in[6];
    const float* W_c1     = (const float*)d_in[7];
    const float* g_c1     = (const float*)d_in[8];
    const float* b_c1     = (const float*)d_in[9];
    const float* W_c2     = (const float*)d_in[10];
    const float* g_n      = (const float*)d_in[11];
    const float* b_n      = (const float*)d_in[12];
    const float* W_m1     = (const float*)d_in[13];
    const float* g_m1     = (const float*)d_in[14];
    const float* bm1      = (const float*)d_in[15];
    const float* W_m2     = (const float*)d_in[16];
    const float* g_m2     = (const float*)d_in[17];
    const float* bm2      = (const float*)d_in[18];
    const int*   hi       = (const int*)d_in[19];
    const int*   wi       = (const int*)d_in[20];

    const int Ntgt = in_sizes[1] / 128;
    const int E    = in_sizes[19];

    char* ws = (char*)d_ws;
    float* acc = (float*)ws;                       // [Ntgt*128] f32 accumulator
    size_t accB = (size_t)Ntgt * 128 * sizeof(float);
    char* blob1 = ws + accB;                       // edge weight blob (S1 bytes)
    char* blob2 = blob1 + S1;                      // node weight blob (S2 bytes)

    (void)hipFuncSetAttribute((const void*)edge_kernel,
        hipFuncAttributeMaxDynamicSharedMemorySize, SMEM_EDGE);
    (void)hipFuncSetAttribute((const void*)node_kernel,
        hipFuncAttributeMaxDynamicSharedMemorySize, SMEM_NODE);

    // ---- prep: zero accumulator, transpose weights to bf16 [N][Kpad] --------
    {
        int n = Ntgt * 128;
        zero_f32<<<(n + 255) / 256, 256, 0, stream>>>(acc, n);
    }
    transpose_bf16<<<(128 * LK1 + 255) / 256, 256, 0, stream>>>(
        W_c1, (__bf16*)(blob1), 256, 128, LK1);
    transpose_bf16<<<(128 * LK2 + 255) / 256, 256, 0, stream>>>(
        W_c2, (__bf16*)(blob1 + OFF_WC2), 128, 128, LK2);
    copy_f32<<<2, 256, 0, stream>>>(W_rp, (float*)(blob1 + OFF_WRP), 512);
    copy_f32<<<1, 128, 0, stream>>>(b_rp, (float*)(blob1 + OFF_BRP), 128);
    copy_f32<<<1, 128, 0, stream>>>(g_c1, (float*)(blob1 + OFF_GC1), 128);
    copy_f32<<<1, 128, 0, stream>>>(b_c1, (float*)(blob1 + OFF_BC1), 128);

    transpose_bf16<<<(128 * LK2 + 255) / 256, 256, 0, stream>>>(
        W_input, (__bf16*)(blob2), 128, 128, LK2);
    transpose_bf16<<<(128 * LK2 + 255) / 256, 256, 0, stream>>>(
        W_m1, (__bf16*)(blob2 + OFF_WM1), 128, 128, LK2);
    transpose_bf16<<<(128 * LK2 + 255) / 256, 256, 0, stream>>>(
        W_m2, (__bf16*)(blob2 + OFF_WM2), 128, 128, LK2);
    copy_f32<<<1, 128, 0, stream>>>(g_n,  (float*)(blob2 + OFF_GN),  128);
    copy_f32<<<1, 128, 0, stream>>>(b_n,  (float*)(blob2 + OFF_BN),  128);
    copy_f32<<<1, 128, 0, stream>>>(g_m1, (float*)(blob2 + OFF_GM1), 128);
    copy_f32<<<1, 128, 0, stream>>>(bm1,  (float*)(blob2 + OFF_BM1), 128);
    copy_f32<<<1, 128, 0, stream>>>(g_m2, (float*)(blob2 + OFF_GM2), 128);
    copy_f32<<<1, 128, 0, stream>>>(bm2,  (float*)(blob2 + OFF_BM2), 128);

    // ---- edge pipeline ------------------------------------------------------
    {
        int ntiles = (E + 15) / 16;
        int grid = 512;                            // 8 waves/block, 1 blk/WGP (LDS)
        int waves = grid * 8;
        int iters = (ntiles + waves - 1) / waves;
        edge_kernel<<<grid, 256, SMEM_EDGE, stream>>>(
            ctx_feat, cpose, tpose, hi, wi, blob1, acc, E, ntiles, iters);
    }
    // ---- node pipeline ------------------------------------------------------
    {
        int ntiles = (Ntgt + 15) / 16;
        int grid = (ntiles + 7) / 8;
        int waves = grid * 8;
        int iters = (ntiles + waves - 1) / waves;
        node_kernel<<<grid, 256, SMEM_NODE, stream>>>(
            tgt_feat, acc, blob2, (float*)d_out, Ntgt, ntiles, iters);
    }
}
